// SimpleCrystalGNN_50972671869258
// MI455X (gfx1250) — matrix operand order repaired and checked by reading, not compiled, not run on hardware
//
#include <hip/hip_runtime.h>
#include <hip/hip_bf16.h>
#include <stdint.h>

typedef __attribute__((ext_vector_type(16))) __bf16 v16bf;
typedef __attribute__((ext_vector_type(8)))  float  v8f;

#define F_DIM   64
#define NBR_DIM 41
#define M_NBR   12
#define KTOT    169   // 2F + NBR
#define NCOLS   128   // 2F
#define ORIG    92
#define KPAD_EMB 96
#define H_DIM   128

__device__ __forceinline__ unsigned short f2bf(float f) {
    unsigned u = __float_as_uint(f);
    u = u + 0x7FFFu + ((u >> 16) & 1u);       // round to nearest even
    return (unsigned short)(u >> 16);
}
__device__ __forceinline__ float softplus_f(float x) {
    return (x > 15.f) ? x : log1pf(__expf(x));
}
__device__ __forceinline__ float sigmoid_f(float x) {
    return 1.f / (1.f + __expf(-x));
}

union AFrag { v16bf v; unsigned u[8]; };
union BFrag { v16bf v; uint4 q[2]; };

// ---------------- weight packing: f32 -> bf16, per-WMMA-fragment layout ---------------
// Fragment layout (B matrix 32x16, 16-bit): lane -> n = lane%16, element s -> k = 8*(lane/16)+s (s<8),
// 16+8*(lane/16)+(s-8) (s>=8).  Stored contiguously per (kstep, coltile): 32 lanes * 32B.
__global__ void prep_weights(const float* __restrict__ Wf, const float* __restrict__ W_emb,
                             unsigned short* __restrict__ WbfFrag,
                             unsigned short* __restrict__ WembFrag) {
    int idx = blockIdx.x * blockDim.x + threadIdx.x;
    const int convTotal = 3 * 6 * 8 * 32 * 16;     // 3 layers, 6 ksteps, 8 coltiles
    if (idx < convTotal) {
        int pos  = idx & 15;
        int lane = (idx >> 4) & 31;
        int ct   = (idx >> 9) & 7;
        int ks   = (idx >> 12) % 6;
        int l    = idx / (6 * 8 * 32 * 16);
        int k = ks * 32 + ((pos >= 8) ? 16 : 0) + ((lane >> 4) << 3) + (pos & 7);
        int n = ct * 16 + (lane & 15);
        float v = (k < KTOT) ? Wf[(l * KTOT + k) * NCOLS + n] : 0.f;
        WbfFrag[idx] = f2bf(v);
    } else {
        int j = idx - convTotal;
        if (j < 3 * 4 * 32 * 16) {                 // 3 ksteps, 4 coltiles for embedding
            int pos  = j & 15;
            int lane = (j >> 4) & 31;
            int ct   = (j >> 9) & 3;
            int ks   = j >> 11;
            int k = ks * 32 + ((pos >= 8) ? 16 : 0) + ((lane >> 4) << 3) + (pos & 7);
            int n = ct * 16 + (lane & 15);
            float v = (k < ORIG) ? W_emb[k * F_DIM + n] : 0.f;
            WembFrag[j] = f2bf(v);
        }
    }
}

// ---------------- embedding: h = atom_fea @ W_emb + b_emb  (WMMA bf16) ----------------
__global__ void __launch_bounds__(128)
embed_gemm(const float* __restrict__ atom_fea, const float* __restrict__ b_emb,
           const unsigned short* __restrict__ WembFrag, float* __restrict__ h_out, int N) {
    __shared__ unsigned short sA[16 * 100];        // 16 rows x 96K, stride 100 (bank pad)
    int tid  = threadIdx.x;
    int lane = tid & 31;
    int w    = tid >> 5;                           // wave = col tile 0..3
    int row0 = blockIdx.x * 16;
    bool full = (row0 + 16 <= N);

    for (int idx = tid; idx < 16 * KPAD_EMB; idx += 128) {
        int r = idx / KPAD_EMB, k = idx % KPAD_EMB;
        int row = row0 + r;
        float v = (k < ORIG && (full || row < N)) ? atom_fea[row * ORIG + k] : 0.f;
        sA[r * 100 + k] = f2bf(v);
    }
    __syncthreads();

    int n0 = w * 16;
    int nc = lane & 15;
    int m  = lane & 15;
    int hi = (lane >> 4) << 3;                     // 0 or 8
    float bias = b_emb[n0 + nc];
    v8f acc;
    #pragma unroll
    for (int i = 0; i < 8; ++i) acc[i] = bias;

    for (int ks = 0; ks < 3; ++ks) {
        AFrag a;
        #pragma unroll
        for (int p = 0; p < 8; ++p) {
            int koff = ((p >= 4) ? 16 : 0) + hi + ((p & 3) << 1);
            a.u[p] = *(const unsigned*)&sA[m * 100 + ks * 32 + koff];
        }
        BFrag b;
        const uint4* bp = (const uint4*)(WembFrag + ((ks * 4 + w) * 32 + lane) * 16);
        b.q[0] = bp[0]; b.q[1] = bp[1];
        acc = __builtin_amdgcn_wmma_f32_16x16x32_bf16(false, a.v, false, b.v,
                                                      (short)0, acc, false, false);
    }
    if (full) {
        #pragma unroll
        for (int v = 0; v < 8; ++v)
            h_out[(row0 + v + hi) * F_DIM + n0 + nc] = acc[v];
    } else {
        #pragma unroll
        for (int v = 0; v < 8; ++v) {
            int row = row0 + v + hi;
            if (row < N) h_out[row * F_DIM + n0 + nc] = acc[v];
        }
    }
}

// ---------------- fused gather + GEMM for one conv layer ------------------------------
// PASS 0: accumulate column sum/sumsq of gated (for BN1 stats), discard output.
// PASS 1: apply BN1 affine, sigmoid(filter)*softplus(core), reduce over 12 neighbors,
//         write summed[N,64] and accumulate BN2 stats.
// K segments align with 32-wide ksteps: ks0-1 = h_self, ks2-3 = h_nbr, ks4-5 = nbr_fea/pad,
// so staging has no per-element segment branches and uses float4 loads for h segments.
template <int PASS>
__global__ void __launch_bounds__(384)
conv_gemm(const float* __restrict__ h_in, const float* __restrict__ nbr_fea,
          const int* __restrict__ nbr_idx, const unsigned short* __restrict__ WbfL,
          const float* __restrict__ bfL, const float* __restrict__ affine1,
          float* __restrict__ stats1, float* __restrict__ summed,
          float* __restrict__ stats2, int N) {
    __shared__ __align__(16) char smem[192 * 64 * 4];  // overlay: A-chunk / act tile
    __shared__ int   sNbr[192];
    __shared__ float sS1[128], sQ1[128];
    __shared__ float sS2[64],  sQ2[64];
    unsigned short* sA   = (unsigned short*)smem;      // 192 rows x 32 k, stride 36
    float*          sAct = (float*)smem;               // 192 rows x 64 cols

    int tid  = threadIdx.x;
    int lane = tid & 31;
    int w    = tid >> 5;                  // wave 0..11 = row tile
    int NM   = N * M_NBR;
    int row0 = blockIdx.x * 192;          // row in [N*M] space
    int i0   = blockIdx.x * 16;           // first atom of this block
    bool full = (row0 + 192 <= NM);       // uniform: whole block in-bounds

    if (PASS == 0) { if (tid < 128) { sS1[tid] = 0.f; sQ1[tid] = 0.f; } }
    else           { if (tid < 64)  { sS2[tid] = 0.f; sQ2[tid] = 0.f; } }
    if (tid < 192) sNbr[tid] = (full || row0 + tid < NM) ? nbr_idx[row0 + tid] : 0;

    int nc = lane & 15;
    int m  = lane & 15;
    int hi = (lane >> 4) << 3;
    v8f acc[8];
    #pragma unroll
    for (int ct = 0; ct < 8; ++ct) {
        float bias = bfL[ct * 16 + nc];
        #pragma unroll
        for (int i = 0; i < 8; ++i) acc[ct][i] = bias;
    }

    for (int ks = 0; ks < 6; ++ks) {
        __syncthreads();
        if (ks < 4) {
            // h segments: 192 rows x 32 cols via float4 + packed bf16 b64 stores
            int kb = (ks & 1) * 32;
            for (int idx = tid; idx < 192 * 8; idx += 384) {
                int r  = idx >> 3;
                int c4 = (idx & 7) << 2;       // 0,4,...,28
                int i;
                if (ks < 2) { i = i0 + r / M_NBR; if (!full && i >= N) i = N - 1; }
                else        { i = sNbr[r]; }
                float4 v = *(const float4*)&h_in[i * F_DIM + kb + c4];
                uint2 pk;
                pk.x = (unsigned)f2bf(v.x) | ((unsigned)f2bf(v.y) << 16);
                pk.y = (unsigned)f2bf(v.z) | ((unsigned)f2bf(v.w) << 16);
                *(uint2*)&sA[r * 36 + c4] = pk;
            }
        } else {
            // nbr_fea segment (41 floats, unaligned rows) + zero pad
            for (int idx = tid; idx < 192 * 32; idx += 384) {
                int kl = idx & 31;
                int r  = idx >> 5;
                int k  = ks * 32 + kl;         // 128..191
                float v = 0.f;
                if (k < KTOT && (full || row0 + r < NM))
                    v = nbr_fea[(row0 + r) * NBR_DIM + (k - 2 * F_DIM)];
                sA[r * 36 + kl] = f2bf(v);
            }
        }
        __syncthreads();
        AFrag a;
        int rbase = (w * 16 + m) * 36;
        #pragma unroll
        for (int p = 0; p < 8; ++p) {
            int koff = ((p >= 4) ? 16 : 0) + hi + ((p & 3) << 1);
            a.u[p] = *(const unsigned*)&sA[rbase + koff];
        }
        #pragma unroll
        for (int ct = 0; ct < 8; ++ct) {
            BFrag b;
            const uint4* bp = (const uint4*)(WbfL + ((ks * 8 + ct) * 32 + lane) * 16);
            b.q[0] = bp[0]; b.q[1] = bp[1];
            acc[ct] = __builtin_amdgcn_wmma_f32_16x16x32_bf16(false, a.v, false, b.v,
                                                              (short)0, acc[ct], false, false);
        }
    }

    if (PASS == 0) {
        #pragma unroll
        for (int ct = 0; ct < 8; ++ct) {
            float s1 = 0.f, s2 = 0.f;
            #pragma unroll
            for (int v = 0; v < 8; ++v) { float x = acc[ct][v]; s1 += x; s2 += x * x; }
            s1 += __shfl_xor(s1, 16, 32);
            s2 += __shfl_xor(s2, 16, 32);
            if (lane < 16) {
                atomicAdd(&sS1[ct * 16 + nc], s1);
                atomicAdd(&sQ1[ct * 16 + nc], s2);
            }
        }
        __syncthreads();
        if (tid < 128) {
            atomicAdd(&stats1[tid],       sS1[tid]);
            atomicAdd(&stats1[128 + tid], sQ1[tid]);
        }
    } else {
        __syncthreads();   // all waves done reading sA before act overlay
        #pragma unroll
        for (int ct = 0; ct < 4; ++ct) {
            int colF = ct * 16 + nc;                 // filter col; core col = colF + 64
            float scF = affine1[colF],      shF = affine1[128 + colF];
            float scC = affine1[colF + 64], shC = affine1[128 + colF + 64];
            #pragma unroll
            for (int v = 0; v < 8; ++v) {
                float f = acc[ct][v]     * scF + shF;
                float c = acc[ct + 4][v] * scC + shC;
                sAct[(w * 16 + v + hi) * 64 + colF] = sigmoid_f(f) * softplus_f(c);
            }
        }
        __syncthreads();
        if (i0 + 16 <= N) {
            for (int o = tid; o < 16 * 64; o += 384) {
                int a_ = o >> 6, c_ = o & 63;
                float s = 0.f;
                #pragma unroll
                for (int j = 0; j < M_NBR; ++j) s += sAct[(a_ * M_NBR + j) * 64 + c_];
                summed[(i0 + a_) * F_DIM + c_] = s;
                atomicAdd(&sS2[c_], s);
                atomicAdd(&sQ2[c_], s * s);
            }
        } else {
            for (int o = tid; o < 16 * 64; o += 384) {
                int a_ = o >> 6, c_ = o & 63;
                float s = 0.f;
                #pragma unroll
                for (int j = 0; j < M_NBR; ++j) s += sAct[(a_ * M_NBR + j) * 64 + c_];
                if (i0 + a_ < N) {
                    summed[(i0 + a_) * F_DIM + c_] = s;
                    atomicAdd(&sS2[c_], s);
                    atomicAdd(&sQ2[c_], s * s);
                }
            }
        }
        __syncthreads();
        if (tid < 64) {
            atomicAdd(&stats2[tid],      sS2[tid]);
            atomicAdd(&stats2[64 + tid], sQ2[tid]);
        }
    }
}

// ---------------- BN finalize: stats -> per-column affine -----------------------------
__global__ void bn_finalize(const float* __restrict__ stats, const float* __restrict__ g,
                            const float* __restrict__ b, float* __restrict__ affine,
                            int C, float invCnt) {
    int c = blockIdx.x * blockDim.x + threadIdx.x;
    if (c < C) {
        float mean = stats[c] * invCnt;
        float var  = stats[C + c] * invCnt - mean * mean;
        float sc   = g[c] * rsqrtf(var + 1e-5f);
        affine[c]     = sc;
        affine[C + c] = b[c] - mean * sc;
    }
}

// ---------------- h_next = softplus(h + bn2(summed)) ----------------------------------
__global__ void update_h(const float* __restrict__ h_in, const float* __restrict__ summed,
                         const float* __restrict__ affine2, float* __restrict__ h_out,
                         int total) {
    int idx = blockIdx.x * blockDim.x + threadIdx.x;
    if (idx < total) {
        int c = idx & 63;
        float s = summed[idx] * affine2[c] + affine2[64 + c];
        h_out[idx] = softplus_f(h_in[idx] + s);
    }
}

// ---------------- segment pooling ------------------------------------------------------
__global__ void zero_f32(float* __restrict__ p, int n) {
    int i = blockIdx.x * blockDim.x + threadIdx.x;
    if (i < n) p[i] = 0.f;
}
__global__ void pool_accum(const float* __restrict__ h, const int* __restrict__ cids,
                           float* __restrict__ psums, float* __restrict__ pcnt, int N) {
    int idx = blockIdx.x * blockDim.x + threadIdx.x;
    if (idx < N * F_DIM) {
        int i = idx >> 6, c = idx & 63;
        int cid = cids[i];
        atomicAdd(&psums[cid * F_DIM + c], h[idx]);
        if (c == 0) atomicAdd(&pcnt[cid], 1.f);
    }
}

// ---------------- head: mean -> fc(softplus) -> out ------------------------------------
__global__ void __launch_bounds__(128)
head_kernel(const float* __restrict__ psums, const float* __restrict__ pcnt,
            const float* __restrict__ W_fc, const float* __restrict__ b_fc,
            const float* __restrict__ W_out, const float* __restrict__ b_out,
            float* __restrict__ out) {
    __shared__ float sc[F_DIM];
    __shared__ float sr[H_DIM];
    int b = blockIdx.x, t = threadIdx.x;
    float cnt = fmaxf(pcnt[b], 1.f);
    if (t < F_DIM) sc[t] = psums[b * F_DIM + t] / cnt;
    __syncthreads();
    float a = b_fc[t];
    for (int j = 0; j < F_DIM; ++j) a += sc[j] * W_fc[j * H_DIM + t];
    sr[t] = softplus_f(a) * W_out[t];
    __syncthreads();
    for (int s = H_DIM / 2; s > 0; s >>= 1) {
        if (t < s) sr[t] += sr[t + s];
        __syncthreads();
    }
    if (t == 0) out[b] = sr[0] + b_out[0];
}

extern "C" void kernel_launch(void* const* d_in, const int* in_sizes, int n_in,
                              void* d_out, int out_size, void* d_ws, size_t ws_size,
                              hipStream_t stream) {
    (void)n_in; (void)ws_size;
    const float* atom_fea = (const float*)d_in[0];
    const float* nbr_fea  = (const float*)d_in[1];
    const int*   nbr_idx  = (const int*)d_in[2];
    const int*   cids     = (const int*)d_in[3];
    // d_in[4] = num_crystals lives on device; out == [B,1] so B == out_size.
    const float* W_emb = (const float*)d_in[5];
    const float* b_emb = (const float*)d_in[6];
    const float* Wf    = (const float*)d_in[7];
    const float* bf    = (const float*)d_in[8];
    const float* g1    = (const float*)d_in[9];
    const float* b1    = (const float*)d_in[10];
    const float* g2    = (const float*)d_in[11];
    const float* b2    = (const float*)d_in[12];
    const float* W_fc  = (const float*)d_in[13];
    const float* b_fc  = (const float*)d_in[14];
    const float* W_out = (const float*)d_in[15];
    const float* b_out = (const float*)d_in[16];
    float* out = (float*)d_out;

    int N  = in_sizes[0] / ORIG;
    int B  = out_size;
    int NM = N * M_NBR;

    char* ws = (char*)d_ws;
    size_t off = 0;
    auto alloc = [&](size_t bytes) { size_t o = off; off = (off + bytes + 255) & ~(size_t)255; return o; };
    float* h0      = (float*)(ws + alloc((size_t)N * F_DIM * 4));
    float* h1      = (float*)(ws + alloc((size_t)N * F_DIM * 4));
    float* summed  = (float*)(ws + alloc((size_t)N * F_DIM * 4));
    unsigned short* WbfFrag  = (unsigned short*)(ws + alloc((size_t)3 * 6 * 8 * 32 * 16 * 2));
    unsigned short* WembFrag = (unsigned short*)(ws + alloc((size_t)3 * 4 * 32 * 16 * 2));
    float* stats1  = (float*)(ws + alloc(256 * 4));
    float* stats2  = (float*)(ws + alloc(128 * 4));
    float* affine1 = (float*)(ws + alloc(256 * 4));
    float* affine2 = (float*)(ws + alloc(128 * 4));
    float* psums   = (float*)(ws + alloc((size_t)B * F_DIM * 4));
    float* pcnt    = (float*)(ws + alloc((size_t)B * 4));

    const int prepTotal = 3 * 6 * 8 * 32 * 16 + 3 * 4 * 32 * 16;
    prep_weights<<<(prepTotal + 255) / 256, 256, 0, stream>>>(Wf, W_emb, WbfFrag, WembFrag);

    embed_gemm<<<(N + 15) / 16, 128, 0, stream>>>(atom_fea, b_emb, WembFrag, h0, N);

    float* hc = h0;
    float* hn = h1;
    int convBlocks = (NM + 191) / 192;
    for (int l = 0; l < 3; ++l) {
        zero_f32<<<2, 256, 0, stream>>>(stats1, 256);
        zero_f32<<<1, 128, 0, stream>>>(stats2, 128);
        const unsigned short* WbfL = WbfFrag + (size_t)l * 6 * 8 * 32 * 16;
        const float* bfL = bf + l * NCOLS;
        conv_gemm<0><<<convBlocks, 384, 0, stream>>>(hc, nbr_fea, nbr_idx, WbfL, bfL,
                                                     affine1, stats1, summed, stats2, N);
        bn_finalize<<<1, 128, 0, stream>>>(stats1, g1 + l * NCOLS, b1 + l * NCOLS,
                                           affine1, 128, 1.f / (float)NM);
        conv_gemm<1><<<convBlocks, 384, 0, stream>>>(hc, nbr_fea, nbr_idx, WbfL, bfL,
                                                     affine1, stats1, summed, stats2, N);
        bn_finalize<<<1, 64, 0, stream>>>(stats2, g2 + l * F_DIM, b2 + l * F_DIM,
                                          affine2, 64, 1.f / (float)N);
        update_h<<<(N * F_DIM + 255) / 256, 256, 0, stream>>>(hc, summed, affine2, hn, N * F_DIM);
        float* t = hc; hc = hn; hn = t;
    }

    zero_f32<<<(B * F_DIM + 255) / 256, 256, 0, stream>>>(psums, B * F_DIM);
    zero_f32<<<(B + 255) / 256, 256, 0, stream>>>(pcnt, B);
    pool_accum<<<(N * F_DIM + 255) / 256, 256, 0, stream>>>(hc, cids, psums, pcnt, N);
    head_kernel<<<B, 128, 0, stream>>>(psums, pcnt, W_fc, b_fc, W_out, b_out, out);
}
